// ConvResidualBlock_3470333575253
// MI455X (gfx1250) — matrix operand order repaired and checked by reading, not compiled, not run on hardware
//
#include <hip/hip_runtime.h>
#include <hip/hip_bf16.h>
#include <math.h>

// ---- problem constants (from reference) ----
#define N_NODES 50000
#define N_EDGES 800000
#define IO_DIM  128
#define HID_DIM 256
#define KQ_DIM  256
#define BN_EPS  1e-5f

typedef __bf16 bf16;
typedef __attribute__((ext_vector_type(16))) __bf16 bf16x16;
typedef __attribute__((ext_vector_type(8)))  __bf16 bf16x8;
typedef __attribute__((ext_vector_type(8)))  float  floatx8;

// =====================================================================
// WMMA GEMM:  C[N,M] = A[N,K] (bf16, row-major) x Bt[M,K] (bf16, row-major
// = B transposed).  One wave computes a 16x(16*CT) super-tile: a single A
// fragment per k-step is reused across CT B-column fragments.  All CT B
// fragments are loaded BEFORE the WMMA chain of each k-step so the
// scheduler can clause the loads and issue WMMAs back-to-back with
// partial s_wait_loadcnt instead of a full drain per WMMA.
// =====================================================================
template <int K, int CT>
__global__ __launch_bounds__(256) void wmma_gemm_bf16(
    const bf16* __restrict__ A, const bf16* __restrict__ Bt,
    float* __restrict__ C, int Nn, int M) {
  const int wave = threadIdx.x >> 5;
  const int lane = threadIdx.x & 31;
  const int tile = blockIdx.x * 8 + wave;
  const int mst  = (M >> 4) / CT;            // column-supertiles across M
  const int tm   = (tile / mst) << 4;
  const int tn0  = (tile % mst) * (CT * 16);
  if (tm >= Nn) return;  // wave-uniform guard: EXEC stays all-ones

  const int l15  = lane & 15;
  const int half = lane >> 4;

  // A fragment rows: lane -> row tm+l15 ; 16 contiguous K from half*16
  const bf16* arow = A  + (size_t)(tm + l15) * K;
  // B fragment cols: lane -> col tn0 + j*16 + l15 ; 16 contiguous K from half*16
  const bf16* bbase = Bt + (size_t)(tn0 + l15) * K + half * 16;

  floatx8 acc[CT];
#pragma unroll
  for (int j = 0; j < CT; ++j) acc[j] = (floatx8){};

#pragma unroll
  for (int kk = 0; kk < K; kk += 32) {
    // ---- issue ALL loads of this k-step first (A + CT B fragments) ----
    bf16x16 a = *(const bf16x16*)(arow + kk + half * 16);
    bf16x16 b[CT];
#pragma unroll
    for (int j = 0; j < CT; ++j)
      b[j] = *(const bf16x16*)(bbase + (size_t)j * 16 * K + kk);



    // ---- WMMA chain: one A fragment feeds CT accumulators ----
#pragma unroll
    for (int j = 0; j < CT; ++j)
      acc[j] = __builtin_amdgcn_wmma_f32_16x16x32_bf16(
          /*neg_a=*/false, a, /*neg_b=*/false, b[j],
          /*c_mod=*/(short)0, acc[j], /*reuse_a=*/false, /*reuse_b=*/false);
  }
  // D layout: VGPR r -> row tm + r + 8*half, col tn0 + j*16 + l15
  float* crow = C + (size_t)(tm + half * 8) * M + tn0 + l15;
#pragma unroll
  for (int j = 0; j < CT; ++j)
#pragma unroll
    for (int r = 0; r < 8; ++r) crow[(size_t)r * M + j * 16] = acc[j][r];
}

// =====================================================================
// helpers
// =====================================================================
__global__ void fill_f32(float* p, float v, size_t n) {
  size_t i = (size_t)blockIdx.x * blockDim.x + threadIdx.x;
  if (i < n) p[i] = v;
}

__global__ void cvt_f32_bf16(const float* __restrict__ x, bf16* __restrict__ y, size_t n) {
  size_t i = (size_t)blockIdx.x * blockDim.x + threadIdx.x;
  if (i < n) y[i] = (bf16)x[i];
}

// W [K,M] f32 row-major  ->  Wt [M,K] bf16 row-major
__global__ void f32_to_bf16_T(const float* __restrict__ W, bf16* __restrict__ Wt, int K, int M) {
  int i = blockIdx.x * blockDim.x + threadIdx.x;
  if (i >= K * M) return;
  int k = i / M, m = i % M;
  Wt[(size_t)m * K + k] = (bf16)W[i];
}

__device__ inline void atomicMaxF(float* addr, float val) {
  unsigned int* p = (unsigned int*)addr;
  unsigned int cur = __float_as_uint(*addr);
  while (__uint_as_float(cur) < val) {
    unsigned int prev = atomicCAS(p, cur, __float_as_uint(val));
    if (prev == cur) break;
    cur = prev;
  }
}

// logits[e] = scale * dot(q[dst[e]], k[src[e]])   (one wave32 per edge)
__global__ void edge_logits_k(const float* __restrict__ q, const float* __restrict__ kmat,
                              const int* __restrict__ src, const int* __restrict__ dst,
                              float* __restrict__ logits, int E, int D, float scale) {
  int e = (int)(((size_t)blockIdx.x * blockDim.x + threadIdx.x) >> 5);
  int lane = threadIdx.x & 31;
  if (e >= E) return;
  const float* qp = q + (size_t)dst[e] * D;
  const float* kp = kmat + (size_t)src[e] * D;
  float acc = 0.f;
  for (int c = lane; c < D; c += 32) acc += qp[c] * kp[c];
#pragma unroll
  for (int off = 16; off; off >>= 1) acc += __shfl_xor(acc, off, 32);
  if (lane == 0) logits[e] = acc * scale;
}

__global__ void seg_max_k(const float* __restrict__ logits, const int* __restrict__ dst,
                          float* __restrict__ m, int E) {
  int e = blockIdx.x * blockDim.x + threadIdx.x;
  if (e < E) atomicMaxF(&m[dst[e]], logits[e]);
}

__global__ void clamp_m_k(float* m, int n) {
  int i = blockIdx.x * blockDim.x + threadIdx.x;
  if (i < n && m[i] < -3.0e38f) m[i] = 0.f;  // non-finite (no in-edges) -> 0
}

// eL[e] = exp(logit - m[dst]);  s[dst] += eL[e]   (eL overwrites logits)
__global__ void exp_sum_k(float* __restrict__ eL, const int* __restrict__ dst,
                          const float* __restrict__ m, float* __restrict__ s, int E) {
  int e = blockIdx.x * blockDim.x + threadIdx.x;
  if (e >= E) return;
  int d = dst[e];
  float ex = expf(eL[e] - m[d]);
  eL[e] = ex;
  atomicAdd(&s[d], ex);
}

// agg[dst] += (e / max(s[dst],1e-16)) * v[src]    (one wave32 per edge)
__global__ void agg_k(const float* __restrict__ eL, const float* __restrict__ s,
                      const int* __restrict__ src, const int* __restrict__ dst,
                      const float* __restrict__ v, float* __restrict__ agg, int E, int D) {
  int e = (int)(((size_t)blockIdx.x * blockDim.x + threadIdx.x) >> 5);
  int lane = threadIdx.x & 31;
  if (e >= E) return;
  int d = dst[e], sn = src[e];
  float w = eL[e] / fmaxf(s[d], 1e-16f);
  const float* vp = v + (size_t)sn * D;
  float* ap = agg + (size_t)d * D;
  for (int c = lane; c < D; c += 32) atomicAdd(&ap[c], w * vp[c]);
}

// h += r + bias[c]
__global__ void combine_k(float* __restrict__ h, const float* __restrict__ r,
                          const float* __restrict__ bias, size_t n, int D) {
  size_t i = (size_t)blockIdx.x * blockDim.x + threadIdx.x;
  if (i < n) h[i] += r[i] + bias[i % D];
}

// per-channel sum / sumsq ; launch <<<256, D>>>
__global__ void bn_stats_k(const float* __restrict__ x, float* __restrict__ sum,
                           float* __restrict__ sq, int Nn, int D) {
  int c = threadIdx.x;
  float s0 = 0.f, s1 = 0.f;
  for (int r = blockIdx.x; r < Nn; r += gridDim.x) {
    float v = x[(size_t)r * D + c];
    s0 += v; s1 += v * v;
  }
  atomicAdd(&sum[c], s0);
  atomicAdd(&sq[c], s1);
}

__global__ void bn_apply_k(const float* __restrict__ x, const float* __restrict__ sum,
                           const float* __restrict__ sq, const float* __restrict__ g,
                           const float* __restrict__ be, const float* __restrict__ residual,
                           float* __restrict__ y, int Nn, int D, int doRelu) {
  size_t i = (size_t)blockIdx.x * blockDim.x + threadIdx.x;
  if (i >= (size_t)Nn * D) return;
  int c = (int)(i % D);
  float mu = sum[c] / Nn;
  float var = sq[c] / Nn - mu * mu;
  float v = (x[i] - mu) * rsqrtf(var + BN_EPS) * g[c] + be[c];
  if (residual) v += residual[i];
  if (doRelu) v = fmaxf(v, 0.f);
  y[i] = v;
}

// =====================================================================
// launcher
// =====================================================================
extern "C" void kernel_launch(void* const* d_in, const int* in_sizes, int n_in,
                              void* d_out, int out_size, void* d_ws, size_t ws_size,
                              hipStream_t stream) {
  (void)in_sizes; (void)n_in; (void)out_size; (void)ws_size;
  const float* x0  = (const float*)d_in[0];
  const int*   edg = (const int*)d_in[1];
  const int*   src = edg;             // edges[0,:]
  const int*   dst = edg + N_EDGES;   // edges[1,:]
  const float* Wq1 = (const float*)d_in[2];
  const float* Wk1 = (const float*)d_in[3];
  const float* Wv1 = (const float*)d_in[4];
  const float* Wr1 = (const float*)d_in[5];
  const float* b1  = (const float*)d_in[6];
  const float* g1  = (const float*)d_in[7];
  const float* be1 = (const float*)d_in[8];
  const float* Wq2 = (const float*)d_in[9];
  const float* Wk2 = (const float*)d_in[10];
  const float* Wv2 = (const float*)d_in[11];
  const float* Wr2 = (const float*)d_in[12];
  const float* b2  = (const float*)d_in[13];
  const float* g2  = (const float*)d_in[14];
  const float* be2 = (const float*)d_in[15];
  float* out = (float*)d_out;

  // ---- workspace carve-up (~183 MB) ----
  char* ws = (char*)d_ws;
  size_t off = 0;
  auto alloc = [&](size_t bytes) { size_t o = off; off += (bytes + 255) & ~(size_t)255; return o; };
  float* bufA = (float*)(ws + alloc((size_t)N_NODES * 256 * 4));  // q, then v
  float* bufB = (float*)(ws + alloc((size_t)N_NODES * 256 * 4));  // k, then r
  float* bufH = (float*)(ws + alloc((size_t)N_NODES * 256 * 4));  // agg/h/x1
  float* eL   = (float*)(ws + alloc((size_t)N_EDGES * 4));
  float* mN   = (float*)(ws + alloc((size_t)N_NODES * 4));
  float* sN   = (float*)(ws + alloc((size_t)N_NODES * 4));
  float* bnS  = (float*)(ws + alloc(256 * 4));
  float* bnQ  = (float*)(ws + alloc(256 * 4));
  bf16*  Xb   = (bf16*)(ws + alloc((size_t)N_NODES * 256 * 2));
  bf16* wq1t = (bf16*)(ws + alloc((size_t)IO_DIM * KQ_DIM * 2));
  bf16* wk1t = (bf16*)(ws + alloc((size_t)IO_DIM * KQ_DIM * 2));
  bf16* wv1t = (bf16*)(ws + alloc((size_t)IO_DIM * HID_DIM * 2));
  bf16* wr1t = (bf16*)(ws + alloc((size_t)IO_DIM * HID_DIM * 2));
  bf16* wq2t = (bf16*)(ws + alloc((size_t)HID_DIM * KQ_DIM * 2));
  bf16* wk2t = (bf16*)(ws + alloc((size_t)HID_DIM * KQ_DIM * 2));
  bf16* wv2t = (bf16*)(ws + alloc((size_t)HID_DIM * IO_DIM * 2));
  bf16* wr2t = (bf16*)(ws + alloc((size_t)HID_DIM * IO_DIM * 2));

  const float scale = 1.0f / 16.0f;  // 1/sqrt(KQ=256)
  auto blocks = [](size_t n, int t) { return (unsigned)((n + t - 1) / t); };

  // ---- weight prep: transpose + bf16 ----
  f32_to_bf16_T<<<blocks(IO_DIM * KQ_DIM, 256), 256, 0, stream>>>(Wq1, wq1t, IO_DIM, KQ_DIM);
  f32_to_bf16_T<<<blocks(IO_DIM * KQ_DIM, 256), 256, 0, stream>>>(Wk1, wk1t, IO_DIM, KQ_DIM);
  f32_to_bf16_T<<<blocks(IO_DIM * HID_DIM, 256), 256, 0, stream>>>(Wv1, wv1t, IO_DIM, HID_DIM);
  f32_to_bf16_T<<<blocks(IO_DIM * HID_DIM, 256), 256, 0, stream>>>(Wr1, wr1t, IO_DIM, HID_DIM);
  f32_to_bf16_T<<<blocks(HID_DIM * KQ_DIM, 256), 256, 0, stream>>>(Wq2, wq2t, HID_DIM, KQ_DIM);
  f32_to_bf16_T<<<blocks(HID_DIM * KQ_DIM, 256), 256, 0, stream>>>(Wk2, wk2t, HID_DIM, KQ_DIM);
  f32_to_bf16_T<<<blocks(HID_DIM * IO_DIM, 256), 256, 0, stream>>>(Wv2, wv2t, HID_DIM, IO_DIM);
  f32_to_bf16_T<<<blocks(HID_DIM * IO_DIM, 256), 256, 0, stream>>>(Wr2, wr2t, HID_DIM, IO_DIM);

  // wave super-tiles: rows/16 * (M/64); 8 waves per block, tail guarded
  const unsigned gemmB_M256 = blocks((size_t)(N_NODES / 16) * (256 / 64), 8);  // 1563
  const unsigned gemmB_M128 = blocks((size_t)(N_NODES / 16) * (128 / 64), 8);  // 782
  const unsigned edgeWaveB = blocks((size_t)N_EDGES * 32, 256);
  const unsigned edgeThrB  = blocks(N_EDGES, 256);

  // ================= layer 1 (in 128 -> hid 256) =================
  cvt_f32_bf16<<<blocks((size_t)N_NODES * IO_DIM, 256), 256, 0, stream>>>(x0, Xb, (size_t)N_NODES * IO_DIM);
  wmma_gemm_bf16<IO_DIM, 4><<<gemmB_M256, 256, 0, stream>>>(Xb, wq1t, bufA, N_NODES, KQ_DIM);
  wmma_gemm_bf16<IO_DIM, 4><<<gemmB_M256, 256, 0, stream>>>(Xb, wk1t, bufB, N_NODES, KQ_DIM);
  edge_logits_k<<<edgeWaveB, 256, 0, stream>>>(bufA, bufB, src, dst, eL, N_EDGES, KQ_DIM, scale);
  wmma_gemm_bf16<IO_DIM, 4><<<gemmB_M256, 256, 0, stream>>>(Xb, wv1t, bufA, N_NODES, HID_DIM);  // v reuses q buf
  wmma_gemm_bf16<IO_DIM, 4><<<gemmB_M256, 256, 0, stream>>>(Xb, wr1t, bufB, N_NODES, HID_DIM);  // r reuses k buf
  fill_f32<<<blocks(N_NODES, 256), 256, 0, stream>>>(mN, -INFINITY, N_NODES);
  fill_f32<<<blocks(N_NODES, 256), 256, 0, stream>>>(sN, 0.f, N_NODES);
  fill_f32<<<blocks((size_t)N_NODES * HID_DIM, 256), 256, 0, stream>>>(bufH, 0.f, (size_t)N_NODES * HID_DIM);
  seg_max_k<<<edgeThrB, 256, 0, stream>>>(eL, dst, mN, N_EDGES);
  clamp_m_k<<<blocks(N_NODES, 256), 256, 0, stream>>>(mN, N_NODES);
  exp_sum_k<<<edgeThrB, 256, 0, stream>>>(eL, dst, mN, sN, N_EDGES);
  agg_k<<<edgeWaveB, 256, 0, stream>>>(eL, sN, src, dst, bufA, bufH, N_EDGES, HID_DIM);
  combine_k<<<blocks((size_t)N_NODES * HID_DIM, 256), 256, 0, stream>>>(bufH, bufB, b1, (size_t)N_NODES * HID_DIM, HID_DIM);
  fill_f32<<<1, 256, 0, stream>>>(bnS, 0.f, 256);
  fill_f32<<<1, 256, 0, stream>>>(bnQ, 0.f, 256);
  bn_stats_k<<<256, HID_DIM, 0, stream>>>(bufH, bnS, bnQ, N_NODES, HID_DIM);
  bn_apply_k<<<blocks((size_t)N_NODES * HID_DIM, 256), 256, 0, stream>>>(
      bufH, bnS, bnQ, g1, be1, nullptr, bufH, N_NODES, HID_DIM, 1);

  // ================= layer 2 (hid 256 -> out 128) =================
  cvt_f32_bf16<<<blocks((size_t)N_NODES * HID_DIM, 256), 256, 0, stream>>>(bufH, Xb, (size_t)N_NODES * HID_DIM);
  wmma_gemm_bf16<HID_DIM, 4><<<gemmB_M256, 256, 0, stream>>>(Xb, wq2t, bufA, N_NODES, KQ_DIM);
  wmma_gemm_bf16<HID_DIM, 4><<<gemmB_M256, 256, 0, stream>>>(Xb, wk2t, bufB, N_NODES, KQ_DIM);
  edge_logits_k<<<edgeWaveB, 256, 0, stream>>>(bufA, bufB, src, dst, eL, N_EDGES, KQ_DIM, scale);
  wmma_gemm_bf16<HID_DIM, 4><<<gemmB_M128, 256, 0, stream>>>(Xb, wv2t, bufA, N_NODES, IO_DIM);
  wmma_gemm_bf16<HID_DIM, 4><<<gemmB_M128, 256, 0, stream>>>(Xb, wr2t, bufB, N_NODES, IO_DIM);
  fill_f32<<<blocks(N_NODES, 256), 256, 0, stream>>>(mN, -INFINITY, N_NODES);
  fill_f32<<<blocks(N_NODES, 256), 256, 0, stream>>>(sN, 0.f, N_NODES);
  fill_f32<<<blocks((size_t)N_NODES * IO_DIM, 256), 256, 0, stream>>>(bufH, 0.f, (size_t)N_NODES * IO_DIM);
  seg_max_k<<<edgeThrB, 256, 0, stream>>>(eL, dst, mN, N_EDGES);
  clamp_m_k<<<blocks(N_NODES, 256), 256, 0, stream>>>(mN, N_NODES);
  exp_sum_k<<<edgeThrB, 256, 0, stream>>>(eL, dst, mN, sN, N_EDGES);
  agg_k<<<edgeWaveB, 256, 0, stream>>>(eL, sN, src, dst, bufA, bufH, N_EDGES, IO_DIM);
  combine_k<<<blocks((size_t)N_NODES * IO_DIM, 256), 256, 0, stream>>>(bufH, bufB, b2, (size_t)N_NODES * IO_DIM, IO_DIM);
  fill_f32<<<1, 256, 0, stream>>>(bnS, 0.f, 256);
  fill_f32<<<1, 256, 0, stream>>>(bnQ, 0.f, 256);
  bn_stats_k<<<256, IO_DIM, 0, stream>>>(bufH, bnS, bnQ, N_NODES, IO_DIM);
  bn_apply_k<<<blocks((size_t)N_NODES * IO_DIM, 256), 256, 0, stream>>>(
      bufH, bnS, bnQ, g2, be2, x0, out, N_NODES, IO_DIM, 1);
}
